// _se_equivariant_net_72730976190791
// MI455X (gfx1250) — matrix operand order repaired and checked by reading, not compiled, not run on hardware
//
#include <hip/hip_runtime.h>

#define NN 512
#define BATCH 2
#define HD 64
#define MD 128
#define CH 32            // neighbors per chunk in edge kernel
#define AHS 136          // Ah row stride (halves), 16B-aligned, bank-rotating
#define MBS 132          // mb row stride (floats), bank-rotating

typedef __attribute__((ext_vector_type(16))) _Float16 v16h;
typedef __attribute__((ext_vector_type(8)))  _Float16 v8h;
typedef __attribute__((ext_vector_type(8)))  float    v8f;

__device__ __forceinline__ float siluf(float x){ return x / (1.0f + __expf(-x)); }
__device__ __forceinline__ float sigmf_(float x){ return 1.0f / (1.0f + __expf(-x)); }

// 16-bit A fragment (16x32), ISA 7.12.2 layout:
// lane L: M = L&15 ; halves 0..7 -> K = kb+0..7, halves 8..15 -> K = 16+kb+0..7, kb = (L>=16)?8:0
__device__ __forceinline__ v16h load_a_frag(const _Float16* A, int ldk, int kt){
  const int lane = threadIdx.x & 31;
  const int m = lane & 15;
  const int kb = (lane >> 4) << 3;
  const _Float16* p = A + m * ldk + kt * 32;
  v8h lo = *(const v8h*)(p + kb);
  v8h hi = *(const v8h*)(p + 16 + kb);
  v16h a;
#pragma unroll
  for (int t = 0; t < 8; ++t){ a[t] = lo[t]; a[8 + t] = hi[t]; }
  return a;
}

// B fragment (32x16) pre-swizzled blob: idx = (((kt*NT)+nt)*32 + lane)*16 + h
__device__ __forceinline__ v16h load_b_frag(const _Float16* blob, int kt, int nt, int NT){
  const int lane = threadIdx.x & 31;
  const _Float16* p = blob + ((size_t)((kt * NT + nt) * 32 + lane)) * 16;
  v8h lo = *(const v8h*)(p);
  v8h hi = *(const v8h*)(p + 8);
  v16h b;
#pragma unroll
  for (int t = 0; t < 8; ++t){ b[t] = lo[t]; b[8 + t] = hi[t]; }
  return b;
}

#define WMMA_F16(a,b,c) __builtin_amdgcn_wmma_f32_16x16x32_f16(false,(a),false,(b),(short)0,(c),false,false)

// ---- weight swizzle: f32 row-major [K x N] (row offset) -> f16 B-fragment blob ----
__global__ __launch_bounds__(256) void swizzle_b_kernel(const float* __restrict__ src,
                                                        int K, int N, int rowOff,
                                                        _Float16* __restrict__ dst){
  const int idx = blockIdx.x * 256 + threadIdx.x;
  if (idx >= K * N) return;
  const int h    = idx & 15;
  const int lane = (idx >> 4) & 31;
  const int tile = idx >> 9;
  const int NT = N >> 4;
  const int nt = tile % NT;
  const int kt = tile / NT;
  const int n = nt * 16 + (lane & 15);
  const int k = kt * 32 + ((h >> 3) << 4) + ((lane >> 4) << 3) + (h & 7);   // B 32x16: h 0..7 -> K kb+0..7, h 8..15 -> K 16+kb+0..7, kb = (lane>=16)?8:0
  dst[idx] = (_Float16)src[(size_t)(rowOff + k) * N + n];
}

// h0 = mean over neighbor-axis of a softmax over that axis == 1/n exactly
__global__ __launch_bounds__(256) void init_h_kernel(float* __restrict__ h){
  const int idx = blockIdx.x * 256 + threadIdx.x;
  if (idx < BATCH * NN * HD) h[idx] = 1.0f / (float)NN;
}

// ---- per-node precompute: a_j = h@W1e[1:65], c_i = h@W1e[65:129] + b1e ----
__global__ __launch_bounds__(256) void node_pre_kernel(
    const float* __restrict__ h,
    const _Float16* __restrict__ WhjSw,   // K=64 N=128 frag blob
    const _Float16* __restrict__ WhiSw,   // K=64 N=128 frag blob
    const float* __restrict__ b1e,
    float* __restrict__ abuf, float* __restrict__ cbuf)
{
  const int b = blockIdx.y;
  const int n0 = blockIdx.x * 16;
  const int tid = threadIdx.x, lane = tid & 31, wave = tid >> 5;
  __shared__ __align__(16) _Float16 Ahh[16 * HD];
  for (int e = tid; e < 16 * HD; e += 256){
    const int r = e >> 6, c = e & 63;
    Ahh[e] = (_Float16)h[((size_t)(b * NN + n0 + r)) * HD + c];
  }
  __syncthreads();
  v16h a0 = load_a_frag(Ahh, HD, 0);
  v16h a1 = load_a_frag(Ahh, HD, 1);
  v8f accA = {}, accC = {};
  accA = WMMA_F16(a0, load_b_frag(WhjSw, 0, wave, 8), accA);
  accA = WMMA_F16(a1, load_b_frag(WhjSw, 1, wave, 8), accA);
  accC = WMMA_F16(a0, load_b_frag(WhiSw, 0, wave, 8), accC);
  accC = WMMA_F16(a1, load_b_frag(WhiSw, 1, wave, 8), accC);
  const int rb = (lane >> 4) << 3;
  const int col = wave * 16 + (lane & 15);
#pragma unroll
  for (int r = 0; r < 8; ++r){
    const size_t gi = ((size_t)(b * NN + n0 + rb + r)) * MD + col;
    abuf[gi] = accA[r];
    cbuf[gi] = accC[r] + b1e[col];
  }
}

// ---- edge kernel: one target node i per block, neighbors in chunks of 32 ----
__global__ __launch_bounds__(256) void egcl_edge_kernel(
    const float* __restrict__ xin,
    const float* __restrict__ abuf,
    const float* __restrict__ cbuf,
    const float* __restrict__ w0,        // phi_e W1 row 0 [128]
    const _Float16* __restrict__ W2eSw,  // 128x128 frag blob
    const float* __restrict__ b2e,
    const float* __restrict__ Winf,      // [128]
    const float* __restrict__ binf,      // [1]
    const float* __restrict__ Wlin0,     // [128]
    const float* __restrict__ W1x,       // [128]
    const float* __restrict__ b1x,       // [128]
    const _Float16* __restrict__ W2xSw,  // 128x128 frag blob
    const float* __restrict__ b2x,
    const float* __restrict__ Wox,       // [128]
    const float* __restrict__ box,       // [1]
    float* __restrict__ xout,
    float* __restrict__ mibuf)
{
  const int b = blockIdx.y;
  const int i = blockIdx.x;
  const int tid = threadIdx.x;
  const int lane = tid & 31;
  const int wave = tid >> 5;
  const int j8 = tid >> 3;   // 0..31 : row for elementwise/dot phases
  const int g8 = tid & 7;    // 0..7  : 16-wide channel group

  __shared__ __align__(16) _Float16 We[16384];   // 32KB: phi_e W2 frags
  __shared__ __align__(16) _Float16 Wx[16384];   // 32KB: phi_x W2 frags
  __shared__ __align__(16) _Float16 Ah[CH * AHS];
  __shared__ float mb[CH * MBS];
  __shared__ float vw0[MD], vci[MD], vb2e[MD], vb2x[MD];
  __shared__ float vWi[MD], vWl[MD], vW1x[MD], vb1x[MD], vWox[MD];
  __shared__ float sqL[CH], rnL[CH], dfL[CH][3], ejL[CH], txL[CH], pxL[CH];
  __shared__ float pA[256], pB[256];

  // ---- async-DMA both weight blobs into LDS (CDNA5 ASYNCcnt path) ----
  {
    const unsigned ldsWe = (unsigned)(size_t)&We[0];  // low 32 bits of generic = LDS byte offset
    const unsigned ldsWx = (unsigned)(size_t)&Wx[0];
#pragma unroll
    for (int k = 0; k < 8; ++k){
      const unsigned off = (unsigned)(tid * 16 + k * 4096);   // 256 thr * 16B * 8 = 32KB
      asm volatile("global_load_async_to_lds_b128 %0, %1, %2"
                   :: "v"(ldsWe + off), "v"(off), "s"(W2eSw) : "memory");
      asm volatile("global_load_async_to_lds_b128 %0, %1, %2"
                   :: "v"(ldsWx + off), "v"(off), "s"(W2xSw) : "memory");
    }
  }

  if (tid < MD){
    vw0[tid]  = w0[tid];
    vci[tid]  = cbuf[((size_t)(b * NN + i)) * MD + tid];
    vb2e[tid] = b2e[tid];
    vb2x[tid] = b2x[tid];
    vWi[tid]  = Winf[tid];
    vWl[tid]  = Wlin0[tid];
    vW1x[tid] = W1x[tid];
    vb1x[tid] = b1x[tid];
    vWox[tid] = Wox[tid];
  }
  const float xi0 = xin[(b * NN + i) * 3 + 0];
  const float xi1 = xin[(b * NN + i) * 3 + 1];
  const float xi2 = xin[(b * NN + i) * 3 + 2];
  const float biv = binf[0];
  const float bov = box[0];

  float macc = 0.0f;   // tid<128: running m_i[tid]
  float sh   = 0.0f;   // tid<3:   running shift component

  asm volatile("s_wait_asynccnt 0" ::: "memory");
  __syncthreads();

  for (int jc = 0; jc < NN / CH; ++jc){
    // distances for this 32-neighbor chunk
    if (tid < CH){
      const int jg = jc * CH + tid;
      float d0 = xin[(b * NN + jg) * 3 + 0] - xi0;
      float d1 = xin[(b * NN + jg) * 3 + 1] - xi1;
      float d2 = xin[(b * NN + jg) * 3 + 2] - xi2;
      if (jg == i){ d0 = d1 = d2 = 0.0f; }
      const float s = d0 * d0 + d1 * d1 + d2 * d2;
      sqL[tid] = s;
      rnL[tid] = 1.0f / (sqrtf(s + 1e-8f) + 1.0f);
      dfL[tid][0] = d0; dfL[tid][1] = d1; dfL[tid][2] = d2;
    }
    __syncthreads();
    // z1 = silu(sq*w0 + a_j + c_i) -> f16 A tile (32 x 128)
    {
      const int jg = jc * CH + j8;
      const float s = sqL[j8];
      const float* ap = abuf + ((size_t)(b * NN + jg)) * MD + g8 * 16;
      _Float16* dst = &Ah[j8 * AHS + g8 * 16];
#pragma unroll
      for (int u = 0; u < 16; ++u){
        const int d = g8 * 16 + u;
        dst[u] = (_Float16)siluf(s * vw0[d] + ap[u] + vci[d]);
      }
    }
    __syncthreads();
    // GEMM1: m = silu(z1 @ W2e + b2e) -- 2 M-tiles per wave, B reused
    {
      v8f acc0 = {}, acc1 = {};
#pragma unroll
      for (int kt = 0; kt < 4; ++kt){
        v16h bb = load_b_frag(We, kt, wave, 8);
        v16h a0 = load_a_frag(Ah, AHS, kt);
        v16h a1 = load_a_frag(Ah + 16 * AHS, AHS, kt);
        acc0 = WMMA_F16(a0, bb, acc0);
        acc1 = WMMA_F16(a1, bb, acc1);
      }
      const int rb = (lane >> 4) << 3;
      const int col = wave * 16 + (lane & 15);
#pragma unroll
      for (int r = 0; r < 8; ++r){
        mb[(r + rb) * MBS + col]        = siluf(acc0[r] + vb2e[col]);
        mb[(16 + r + rb) * MBS + col]   = siluf(acc1[r] + vb2e[col]);
      }
    }
    __syncthreads();
    // per-edge dots: t_inf = m.Winf, t_x = m.Wlin0
    {
      float pi = 0.0f, pl = 0.0f;
      const float* mr = mb + j8 * MBS + g8 * 16;
#pragma unroll
      for (int u = 0; u < 16; ++u){
        const float m = mr[u]; const int d = g8 * 16 + u;
        pi += m * vWi[d]; pl += m * vWl[d];
      }
      pA[tid] = pi; pB[tid] = pl;
    }
    __syncthreads();
    if (tid < CH){
      float si = 0.0f, sl = 0.0f;
#pragma unroll
      for (int u = 0; u < 8; ++u){ si += pA[tid * 8 + u]; sl += pB[tid * 8 + u]; }
      const int jg = jc * CH + tid;
      const float mask = (jg == i) ? 0.0f : 1.0f;
      ejL[tid] = mask * sigmf_(si + biv);
      txL[tid] = mask * sl;
    }
    __syncthreads();
    // m_i += e_ij * m_ij  (e==0 on diagonal)
    if (tid < MD){
      float a2 = macc;
#pragma unroll
      for (int j = 0; j < CH; ++j) a2 += ejL[j] * mb[j * MBS + tid];
      macc = a2;
    }
    // z2 = silu(t_x * W1x + b1x) -> f16 A tile
    {
      const float t = txL[j8];
      _Float16* dst = &Ah[j8 * AHS + g8 * 16];
#pragma unroll
      for (int u = 0; u < 16; ++u){
        const int d = g8 * 16 + u;
        dst[u] = (_Float16)siluf(t * vW1x[d] + vb1x[d]);
      }
    }
    __syncthreads();
    // GEMM2: u = silu(z2 @ W2x + b2x)
    {
      v8f acc0 = {}, acc1 = {};
#pragma unroll
      for (int kt = 0; kt < 4; ++kt){
        v16h bb = load_b_frag(Wx, kt, wave, 8);
        v16h a0 = load_a_frag(Ah, AHS, kt);
        v16h a1 = load_a_frag(Ah + 16 * AHS, AHS, kt);
        acc0 = WMMA_F16(a0, bb, acc0);
        acc1 = WMMA_F16(a1, bb, acc1);
      }
      const int rb = (lane >> 4) << 3;
      const int col = wave * 16 + (lane & 15);
#pragma unroll
      for (int r = 0; r < 8; ++r){
        mb[(r + rb) * MBS + col]      = siluf(acc0[r] + vb2x[col]);
        mb[(16 + r + rb) * MBS + col] = siluf(acc1[r] + vb2x[col]);
      }
    }
    __syncthreads();
    // phi_x = 2*tanh(u . Wox + box), masked
    {
      float po = 0.0f;
      const float* mr = mb + j8 * MBS + g8 * 16;
#pragma unroll
      for (int u = 0; u < 16; ++u) po += mr[u] * vWox[g8 * 16 + u];
      pA[tid] = po;
    }
    __syncthreads();
    if (tid < CH){
      float so = 0.0f;
#pragma unroll
      for (int u = 0; u < 8; ++u) so += pA[tid * 8 + u];
      const int jg = jc * CH + tid;
      const float mask = (jg == i) ? 0.0f : 1.0f;
      pxL[tid] = mask * 2.0f * tanhf(so + bov);
    }
    __syncthreads();
    if (tid < 3){
      float s2 = sh;
#pragma unroll
      for (int j = 0; j < CH; ++j) s2 += dfL[j][tid] * rnL[j] * pxL[j];
      sh = s2;
    }
    __syncthreads();
  }

  if (tid < MD) mibuf[((size_t)(b * NN + i)) * MD + tid] = macc;
  if (tid < 3){
    const float xi = (tid == 0) ? xi0 : ((tid == 1) ? xi1 : xi2);
    xout[(b * NN + i) * 3 + tid] = xi + sh * (1.0f / (float)(NN - 1));
  }
}

// ---- per-node h update: h += phi_h_out( phi_h_mlp([m_i, h]) ) ----
__global__ __launch_bounds__(256) void node_h_kernel(
    const float* __restrict__ hin,
    const float* __restrict__ mibuf,
    const _Float16* __restrict__ Wh1Sw,  // K=192 N=128
    const float* __restrict__ b1h,
    const _Float16* __restrict__ Wh2Sw,  // K=128 N=128
    const float* __restrict__ b2h,
    const _Float16* __restrict__ WhoSw,  // K=128 N=64
    const float* __restrict__ bho,
    float* __restrict__ hout)
{
  const int b = blockIdx.y;
  const int n0 = blockIdx.x * 16;
  const int tid = threadIdx.x, lane = tid & 31, wave = tid >> 5;
  __shared__ __align__(16) _Float16 A1[16 * 192];
  __shared__ __align__(16) _Float16 A2[16 * MD];
  for (int e = tid; e < 16 * 192; e += 256){
    const int r = e / 192, c = e % 192;
    const float v = (c < MD) ? mibuf[((size_t)(b * NN + n0 + r)) * MD + c]
                             : hin[((size_t)(b * NN + n0 + r)) * HD + (c - MD)];
    A1[e] = (_Float16)v;
  }
  __syncthreads();
  const int rb = (lane >> 4) << 3;
  { // GEMM1 K=192, silu
    v8f acc = {};
#pragma unroll
    for (int kt = 0; kt < 6; ++kt)
      acc = WMMA_F16(load_a_frag(A1, 192, kt), load_b_frag(Wh1Sw, kt, wave, 8), acc);
    const int col = wave * 16 + (lane & 15);
#pragma unroll
    for (int r = 0; r < 8; ++r)
      A2[(r + rb) * MD + col] = (_Float16)siluf(acc[r] + b1h[col]);
  }
  __syncthreads();
  { // GEMM2 K=128, no activation
    v8f acc = {};
#pragma unroll
    for (int kt = 0; kt < 4; ++kt)
      acc = WMMA_F16(load_a_frag(A2, MD, kt), load_b_frag(Wh2Sw, kt, wave, 8), acc);
    const int col = wave * 16 + (lane & 15);
#pragma unroll
    for (int r = 0; r < 8; ++r)
      A1[(r + rb) * MD + col] = (_Float16)(acc[r] + b2h[col]);
  }
  __syncthreads();
  if (wave < 4){ // GEMM3 K=128 N=64, residual add
    v8f acc = {};
#pragma unroll
    for (int kt = 0; kt < 4; ++kt)
      acc = WMMA_F16(load_a_frag(A1, MD, kt), load_b_frag(WhoSw, kt, wave, 4), acc);
    const int col = wave * 16 + (lane & 15);
#pragma unroll
    for (int r = 0; r < 8; ++r){
      const size_t gi = ((size_t)(b * NN + n0 + rb + r)) * HD + col;
      hout[gi] = hin[gi] + acc[r] + bho[col];
    }
  }
}

extern "C" void kernel_launch(void* const* d_in, const int* in_sizes, int n_in,
                              void* d_out, int out_size, void* d_ws, size_t ws_size,
                              hipStream_t stream)
{
  (void)in_sizes; (void)n_in; (void)out_size; (void)ws_size;
  const float* x_in = (const float*)d_in[0];
  // pytree flatten order: x, embed(W,b), then per layer (keys sorted):
  // phi_e(W1,b1,W2,b2), phi_h_mlp(W1,b1,W2,b2), phi_h_out(W,b),
  // phi_inf(W,b), phi_x_lin0, phi_x_mlp(W1,b1,W2,b2), phi_x_out(W,b)  -> 19/layer
  auto P = [&](int i) -> const float* { return (const float*)d_in[i]; };

  char* wsp = (char*)d_ws;
  auto carve = [&](size_t bytes) -> void* {
    void* p = (void*)wsp; wsp += (bytes + 255) & ~(size_t)255; return p;
  };

  float* xb0   = (float*)carve((size_t)BATCH * NN * 3 * sizeof(float));
  float* xb1   = (float*)carve((size_t)BATCH * NN * 3 * sizeof(float));
  float* hb0   = (float*)carve((size_t)BATCH * NN * HD * sizeof(float));
  float* hb1   = (float*)carve((size_t)BATCH * NN * HD * sizeof(float));
  float* abuf  = (float*)carve((size_t)BATCH * NN * MD * sizeof(float));
  float* cbuf  = (float*)carve((size_t)BATCH * NN * MD * sizeof(float));
  float* mibuf = (float*)carve((size_t)BATCH * NN * MD * sizeof(float));
  _Float16 *sW2e[3], *sW2x[3], *sWhj[3], *sWhi[3], *sWh1[3], *sWh2[3], *sWho[3];
  for (int l = 0; l < 3; ++l){
    sW2e[l] = (_Float16*)carve(128 * 128 * sizeof(_Float16));
    sW2x[l] = (_Float16*)carve(128 * 128 * sizeof(_Float16));
    sWhj[l] = (_Float16*)carve(64 * 128 * sizeof(_Float16));
    sWhi[l] = (_Float16*)carve(64 * 128 * sizeof(_Float16));
    sWh1[l] = (_Float16*)carve(192 * 128 * sizeof(_Float16));
    sWh2[l] = (_Float16*)carve(128 * 128 * sizeof(_Float16));
    sWho[l] = (_Float16*)carve(128 * 64 * sizeof(_Float16));
  }

  hipMemcpyAsync(xb0, x_in, (size_t)BATCH * NN * 3 * sizeof(float),
                 hipMemcpyDeviceToDevice, stream);
  init_h_kernel<<<(BATCH * NN * HD + 255) / 256, 256, 0, stream>>>(hb0);

  auto swz = [&](const float* src, int K, int N, int rowOff, _Float16* dst){
    swizzle_b_kernel<<<(K * N + 255) / 256, 256, 0, stream>>>(src, K, N, rowOff, dst);
  };
  for (int l = 0; l < 3; ++l){
    const int base = 3 + 19 * l;
    swz(P(base + 2),  128, 128, 0,  sW2e[l]);   // phi_e W2
    swz(P(base + 15), 128, 128, 0,  sW2x[l]);   // phi_x_mlp W2
    swz(P(base + 0),  64,  128, 1,  sWhj[l]);   // phi_e W1 rows 1..64 (h_j part)
    swz(P(base + 0),  64,  128, 65, sWhi[l]);   // phi_e W1 rows 65..128 (h_i part)
    if (l < 2){
      swz(P(base + 4), 192, 128, 0, sWh1[l]);   // phi_h_mlp W1
      swz(P(base + 6), 128, 128, 0, sWh2[l]);   // phi_h_mlp W2
      swz(P(base + 8), 128, 64,  0, sWho[l]);   // phi_h_out W
    }
  }

  float* xcur = xb0;
  float* hcur = hb0;
  for (int l = 0; l < 3; ++l){
    const int base = 3 + 19 * l;
    node_pre_kernel<<<dim3(NN / 16, BATCH), 256, 0, stream>>>(
        hcur, sWhj[l], sWhi[l], P(base + 1), abuf, cbuf);
    float* xnext = (l == 2) ? (float*)d_out : ((l == 0) ? xb1 : xb0);
    egcl_edge_kernel<<<dim3(NN, BATCH), 256, 0, stream>>>(
        xcur, abuf, cbuf,
        P(base + 0), sW2e[l], P(base + 3),
        P(base + 10), P(base + 11), P(base + 12),
        P(base + 13), P(base + 14), sW2x[l], P(base + 16),
        P(base + 17), P(base + 18),
        xnext, mibuf);
    if (l < 2){
      float* hnext = (l == 0) ? hb1 : hb0;
      node_h_kernel<<<dim3(NN / 16, BATCH), 256, 0, stream>>>(
          hcur, mibuf, sWh1[l], P(base + 5), sWh2[l], P(base + 7),
          sWho[l], P(base + 9), hnext);
      hcur = hnext;
    }
    xcur = xnext;
  }
}
